// FactorGraphConvolution_77704548319642
// MI455X (gfx1250) — matrix operand order repaired and checked by reading, not compiled, not run on hardware
//
#include <hip/hip_runtime.h>
#include <hip/hip_bf16.h>

// ---------------------------------------------------------------------------
// FactorGraphConvolution for MI455X (gfx1250, wave32, WMMA)
// N = 12288, F_IN = F_OUT = 128
//
// Pipeline:
//   1) transpose_to_bf16: feats -> featsT bf16 [128][12288]; weights -> [n][k] bf16
//   2) fgc_spmm: stream node_adj/edge_adj once (non-temporal); 3 big GEMMs via
//      v_wmma_f32_16x16x32_bf16; 0/1 masks computed on the fly in registers.
//   3) fgc_combine: small GEMMs (K=256 node + K=128 edge) via WMMA + bias/relu.
// ---------------------------------------------------------------------------

#define NN 12288
#define FF 128

typedef __attribute__((ext_vector_type(16))) __bf16 v16bf;
typedef __attribute__((ext_vector_type(8)))  float  v8f;
typedef __attribute__((ext_vector_type(2)))  float  v2f;

union BV32 { uint4 u[2]; v16bf v; };   // 32 bytes: one WMMA 16-bf16 operand

// ---------------------------------------------------------------------------
// Generic tiled transpose with f32 -> bf16 conversion: dst[c*R + r] = src[r*C + c]
// ---------------------------------------------------------------------------
__global__ __launch_bounds__(256)
void transpose_to_bf16(const float* __restrict__ src, __bf16* __restrict__ dst,
                       int R, int C) {
    __shared__ float tile[32][33];
    const int r0 = blockIdx.x * 32;
    const int c0 = blockIdx.y * 32;
    const int tx = threadIdx.x;   // 0..31
    const int ty = threadIdx.y;   // 0..7
    #pragma unroll
    for (int i = 0; i < 32; i += 8) {
        int r = r0 + ty + i, c = c0 + tx;
        if (r < R && c < C) tile[ty + i][tx] = src[(size_t)r * C + c];
    }
    __syncthreads();
    #pragma unroll
    for (int i = 0; i < 32; i += 8) {
        int c = c0 + ty + i, r = r0 + tx;
        if (r < R && c < C) dst[(size_t)c * R + r] = (__bf16)tile[tx][ty + i];
    }
}

// ---------------------------------------------------------------------------
// Main kernel: pos_sup = (node_adj>0)@feats, neg_sup = (node_adj<0)@feats,
//              edge_sup = edge_adj@feats.  Streams the two 604MB matrices once.
//
// Block = 128 threads (4 waves), covering 32 rows (2 m-tiles of 16).
// Wave w: m-tile = w>>1, columns [ (w&1)*64 , +64 )  = four 16-col n-tiles.
// All four B fragments are loaded in one clause before the 12 WMMAs so the
// loads for tiles c+1.. overlap the WMMAs for tile c (partial loadcnt waits).
// ---------------------------------------------------------------------------
__global__ __launch_bounds__(128)
void fgc_spmm(const float* __restrict__ node_adj,
              const float* __restrict__ edge_adj,
              const __bf16* __restrict__ featsT,   // [128][12288] bf16
              __bf16* __restrict__ posS,           // [12288][128] bf16
              __bf16* __restrict__ negS,
              __bf16* __restrict__ edgeS) {
    const int tid  = threadIdx.x;
    const int wave = tid >> 5;
    const int lane = tid & 31;
    const int lr   = lane & 15;      // row within m-tile (A) / col within n-tile (B,D)
    const int kh   = lane >> 4;      // K-half selector
    const int mBase = blockIdx.x * 32 + (wave >> 1) * 16;
    const int cb    = (wave & 1) * 64;

    const float*  nrow = node_adj + (size_t)(mBase + lr) * NN;
    const float*  erow = edge_adj + (size_t)(mBase + lr) * NN;
    const __bf16* bp[4];
    #pragma unroll
    for (int c = 0; c < 4; ++c)
        bp[c] = featsT + (size_t)(cb + 16 * c + lr) * NN;

    v8f accP[4] = {}, accN[4] = {}, accE[4] = {};

    const __bf16 ONE  = (__bf16)1.0f;
    const __bf16 ZERO = (__bf16)0.0f;

    for (int k0 = 0; k0 < NN; k0 += 32) {
        // Prefetch the adjacency cache line 8 K-steps ahead (one 128B line/step).
        __builtin_prefetch(nrow + k0 + 256, 0, 1);
        __builtin_prefetch(erow + k0 + 256, 0, 1);

        // ---- B fragments first: 4 tiles, one 8-load clause (L2-resident featsT).
        //      Contiguous K order (same as A): lane(lr,kh), VGPR j holds K = 16*kh + 2j
        //      -> 16 contiguous bf16 per fragment.
        const int kb = k0 + 16 * kh;
        BV32 b[4];
        #pragma unroll
        for (int c = 0; c < 4; ++c) {
            b[c].u[0] = *(const uint4*)(bp[c] + kb);
            b[c].u[1] = *(const uint4*)(bp[c] + kb + 8);
        }

        // ---- A fragments: 16x32 tiles. Contiguous K order (same as B): lane(lr,kh), VGPR j holds
        //      K = 2j + 16*kh  (pairs packed in one dword).
        //      Adjacency is streamed once -> non-temporal loads (keep L2 for featsT).
        v16bf ap, an, ae;
        #pragma unroll
        for (int j = 0; j < 8; ++j) {
            const int kj = k0 + 2 * j + 16 * kh;
            const v2f aN = __builtin_nontemporal_load((const v2f*)(nrow + kj));
            const v2f aE = __builtin_nontemporal_load((const v2f*)(erow + kj));
            ap[2 * j]     = (aN.x > 0.0f) ? ONE : ZERO;
            ap[2 * j + 1] = (aN.y > 0.0f) ? ONE : ZERO;
            an[2 * j]     = (aN.x < 0.0f) ? ONE : ZERO;
            an[2 * j + 1] = (aN.y < 0.0f) ? ONE : ZERO;
            ae[2 * j]     = (__bf16)aE.x;
            ae[2 * j + 1] = (__bf16)aE.y;
        }

        // ---- 12 WMMAs; B loads for later tiles overlap WMMAs of earlier tiles.
        #pragma unroll
        for (int c = 0; c < 4; ++c) {
            accP[c] = __builtin_amdgcn_wmma_f32_16x16x32_bf16(false, ap, false, b[c].v, (short)0, accP[c], false, false);
            accN[c] = __builtin_amdgcn_wmma_f32_16x16x32_bf16(false, an, false, b[c].v, (short)0, accN[c], false, false);
            accE[c] = __builtin_amdgcn_wmma_f32_16x16x32_bf16(false, ae, false, b[c].v, (short)0, accE[c], false, false);
        }
    }

    // D layout: lane -> col n = lr (+16*c), VGPR j -> row m = j + 8*kh
    #pragma unroll
    for (int c = 0; c < 4; ++c) {
        #pragma unroll
        for (int j = 0; j < 8; ++j) {
            const int row = mBase + j + 8 * kh;
            const size_t o = (size_t)row * FF + cb + 16 * c + lr;
            posS[o]  = (__bf16)accP[c][j];
            negS[o]  = (__bf16)accN[c][j];
            edgeS[o] = (__bf16)accE[c][j];
        }
    }
}

// ---------------------------------------------------------------------------
// Combine: out = relu([posS,negS] @ node_weight + nb) + edgeS @ edge_weight + eb
// Block = 256 threads (8 waves) = one 16-row m-tile; wave w = n-tile w.
// ---------------------------------------------------------------------------
__global__ __launch_bounds__(256)
void fgc_combine(const __bf16* __restrict__ posS,
                 const __bf16* __restrict__ negS,
                 const __bf16* __restrict__ edgeS,
                 const __bf16* __restrict__ nodeWt,   // [n=128][k=256] bf16
                 const __bf16* __restrict__ edgeWt,   // [n=128][k=128] bf16
                 const float* __restrict__ node_bias,
                 const float* __restrict__ edge_bias,
                 float* __restrict__ out) {
    const int tid  = threadIdx.x;
    const int wave = tid >> 5;
    const int lane = tid & 31;
    const int lr   = lane & 15;
    const int kh   = lane >> 4;
    const int m0   = blockIdx.x * 16;
    const int col  = wave * 16 + lr;

    const __bf16* prow = posS  + (size_t)(m0 + lr) * FF;
    const __bf16* qrow = negS  + (size_t)(m0 + lr) * FF;
    const __bf16* erow = edgeS + (size_t)(m0 + lr) * FF;
    const __bf16* wn   = nodeWt + (size_t)col * 256;
    const __bf16* we   = edgeWt + (size_t)col * FF;

    v8f accN = {}, accE = {};

    // node, pos half (W rows 0..127)
    #pragma unroll
    for (int k0 = 0; k0 < 128; k0 += 32) {
        BV32 a, b;
        a.u[0] = *(const uint4*)(prow + k0 + 16 * kh);
        a.u[1] = *(const uint4*)(prow + k0 + 16 * kh + 8);
        b.u[0] = *(const uint4*)(wn + k0 + 16 * kh);
        b.u[1] = *(const uint4*)(wn + k0 + 16 * kh + 8);
        accN = __builtin_amdgcn_wmma_f32_16x16x32_bf16(false, a.v, false, b.v, (short)0, accN, false, false);
    }
    // node, neg half (W rows 128..255)
    #pragma unroll
    for (int k0 = 0; k0 < 128; k0 += 32) {
        BV32 a, b;
        a.u[0] = *(const uint4*)(qrow + k0 + 16 * kh);
        a.u[1] = *(const uint4*)(qrow + k0 + 16 * kh + 8);
        b.u[0] = *(const uint4*)(wn + 128 + k0 + 16 * kh);
        b.u[1] = *(const uint4*)(wn + 128 + k0 + 16 * kh + 8);
        accN = __builtin_amdgcn_wmma_f32_16x16x32_bf16(false, a.v, false, b.v, (short)0, accN, false, false);
    }
    // edge
    #pragma unroll
    for (int k0 = 0; k0 < 128; k0 += 32) {
        BV32 a, b;
        a.u[0] = *(const uint4*)(erow + k0 + 16 * kh);
        a.u[1] = *(const uint4*)(erow + k0 + 16 * kh + 8);
        b.u[0] = *(const uint4*)(we + k0 + 16 * kh);
        b.u[1] = *(const uint4*)(we + k0 + 16 * kh + 8);
        accE = __builtin_amdgcn_wmma_f32_16x16x32_bf16(false, a.v, false, b.v, (short)0, accE, false, false);
    }

    const float nb = node_bias[col];
    const float eb = edge_bias[col];
    #pragma unroll
    for (int j = 0; j < 8; ++j) {
        const int row = m0 + j + 8 * kh;
        out[(size_t)row * FF + col] = fmaxf(accN[j] + nb, 0.0f) + accE[j] + eb;
    }
}

// ---------------------------------------------------------------------------
extern "C" void kernel_launch(void* const* d_in, const int* in_sizes, int n_in,
                              void* d_out, int out_size, void* d_ws, size_t ws_size,
                              hipStream_t stream) {
    (void)in_sizes; (void)n_in; (void)out_size; (void)ws_size;

    const float* feats    = (const float*)d_in[0];   // [12288][128]
    const float* node_adj = (const float*)d_in[1];   // [12288][12288]
    const float* edge_adj = (const float*)d_in[2];   // [12288][12288]
    const float* node_w   = (const float*)d_in[3];   // [256][128]
    const float* node_b   = (const float*)d_in[4];   // [128]
    const float* edge_w   = (const float*)d_in[5];   // [128][128]
    const float* edge_b   = (const float*)d_in[6];   // [128]
    float* out = (float*)d_out;                      // [12288][128]

    // Workspace layout (all bf16): featsT | nodeWt | edgeWt | posS | negS | edgeS
    __bf16* featsT = (__bf16*)d_ws;                          // 128*12288
    __bf16* nodeWt = featsT + (size_t)FF * NN;               // 128*256
    __bf16* edgeWt = nodeWt + (size_t)FF * 256;              // 128*128
    __bf16* posS   = edgeWt + (size_t)FF * FF;               // 12288*128
    __bf16* negS   = posS   + (size_t)NN * FF;
    __bf16* edgeS  = negS   + (size_t)NN * FF;

    dim3 tb(32, 8);
    transpose_to_bf16<<<dim3(NN / 32, FF / 32), tb, 0, stream>>>(feats,  featsT, NN,  FF);
    transpose_to_bf16<<<dim3(256 / 32, FF / 32), tb, 0, stream>>>(node_w, nodeWt, 256, FF);
    transpose_to_bf16<<<dim3(FF / 32, FF / 32), tb, 0, stream>>>(edge_w, edgeWt, FF,  FF);

    fgc_spmm<<<NN / 32, 128, 0, stream>>>(node_adj, edge_adj, featsT, posS, negS, edgeS);

    fgc_combine<<<NN / 16, 256, 0, stream>>>(posS, negS, edgeS, nodeWt, edgeWt,
                                             node_b, edge_b, out);
}
